// Memory_Unit_67671504715869
// MI455X (gfx1250) — matrix operand order repaired and checked by reading, not compiled, not run on hardware
//
#include <hip/hip_runtime.h>
#include <hip/hip_bf16.h>
#include <cstdint>

#define NUMS  512
#define DIM   1024
#define MROWS 32768   // B*T = 16*2048
#define KTOP  33

typedef __bf16 bf16_t;
typedef __bf16 v16bf __attribute__((ext_vector_type(16)));
typedef __bf16 v8bf  __attribute__((ext_vector_type(8)));
typedef float  v8f   __attribute__((ext_vector_type(8)));

typedef unsigned int u32x4 __attribute__((ext_vector_type(4)));
typedef int          i32x4 __attribute__((ext_vector_type(4)));
typedef int          i32x8 __attribute__((ext_vector_type(8)));

#define ATT_STRIDE 528       // 512 + 16 pad: rows stay 32B-aligned
#define STG_STRIDE 40        // bf16 elems per staged row: 32 data + 8 pad = 80 B

// LDS byte offset of a __shared__ pointer (flat LDS-aperture address, low 32 bits)
__device__ __forceinline__ uint32_t lds_off(const void* p) {
    return (uint32_t)(uintptr_t)p;
}

// Issue one TDM 2D tile load: tile_d0 dwords x tile_d1 rows from a row-major
// tensor (row stride = stride0 dwords), into LDS with 4-dword padding after
// every 16 dwords (-> 80B LDS row stride). Descriptor per CDNA5 ISA 8.3/8.4.
__device__ __forceinline__ void tdm_load_tile(const void* gaddr, uint32_t ldsaddr,
                                              unsigned tensor_d0, unsigned tensor_d1,
                                              unsigned tile_d0, unsigned tile_d1,
                                              unsigned stride0) {
    uint64_t ga = (uint64_t)(uintptr_t)gaddr;
    u32x4 g0;
    g0[0] = 1u;                                               // count=1, user mode
    g0[1] = ldsaddr;                                          // lds_addr
    g0[2] = (uint32_t)ga;                                     // global_addr[31:0]
    g0[3] = (uint32_t)((ga >> 32) & 0x1FFFFFFu) | (2u << 30); // addr[56:32] | type=2
    i32x8 g1;
    // data_size=4B(code 2) | pad_enable | pad_interval=16dw(3) | pad_amount=4dw(3)
    g1[0] = (2 << 16) | (1 << 20) | (3 << 22) | (3 << 25);
    g1[1] = (int)((tensor_d0 & 0xFFFFu) << 16);               // tensor_dim0 lo16
    g1[2] = (int)(((tensor_d0 >> 16) & 0xFFFFu) | ((tensor_d1 & 0xFFFFu) << 16));
    g1[3] = (int)(((tensor_d1 >> 16) & 0xFFFFu) | (tile_d0 << 16));
    g1[4] = (int)tile_d1;                                     // tile_dim1 (tile_dim2=0)
    g1[5] = (int)stride0;                                     // tensor_dim0_stride lo32
    g1[6] = 0;                                                // stride0 hi16 | stride1 lo16
    g1[7] = 0;                                                // stride1 hi32
    i32x4 g2 = {0, 0, 0, 0};
    i32x4 g3 = {0, 0, 0, 0};
#if defined(__clang_major__) && (__clang_major__ >= 23)
    i32x8 g4 = {0, 0, 0, 0, 0, 0, 0, 0};
    __builtin_amdgcn_tensor_load_to_lds(g0, g1, g2, g3, g4, 0);
#else
    __builtin_amdgcn_tensor_load_to_lds(g0, g1, g2, g3, 0);
#endif
}

// Pre-convert memory_block f32 -> bf16 (row major) and bf16 transposed (for GEMM2 NN).
__global__ __launch_bounds__(256) void memunit_prep(const float* __restrict__ mem,
                                                    bf16_t* __restrict__ mbf,
                                                    bf16_t* __restrict__ mbfT) {
    int idx = blockIdx.x * 256 + threadIdx.x;   // 0 .. NUMS*DIM-1
    int row = idx >> 10;
    int col = idx & (DIM - 1);
    bf16_t b = (bf16_t)mem[idx];
    mbf[idx] = b;
    mbfT[(size_t)col * NUMS + row] = b;
}

__global__ __launch_bounds__(256) void memunit_main(const float* __restrict__ data,
                                                    const bf16_t* __restrict__ mbf,
                                                    const bf16_t* __restrict__ mbfT,
                                                    float* __restrict__ out_t,
                                                    float* __restrict__ out_aug) {
    __shared__ __align__(128) bf16_t att[64][ATT_STRIDE];            // 67584 B
    __shared__ __align__(128) bf16_t stage[2][512 * STG_STRIDE];     // 2 x 40960 B

    const int tid   = threadIdx.x;
    const int lane  = tid & 31;
    const int wave  = tid >> 5;
    const int c     = lane & 15;   // fragment column / row-in-strip
    const int half  = lane >> 4;   // lane half (WMMA layout)
    const int strip = wave & 3;    // which 16-row strip of the 64-row tile
    const int nsel  = wave >> 2;   // which column half

    const int row0 = blockIdx.x * 64;
    const uint32_t stg_base[2] = { lds_off(&stage[0][0]), lds_off(&stage[1][0]) };

    // load one 16x32 bf16 B fragment from a stage buffer (staged row = lr)
    auto load_bfrag = [&](int buf, int nc) -> v16bf {
        const int lr = nsel * 256 + nc * 16 + c;
        const v8bf blo = *(const v8bf*)&stage[buf][lr * STG_STRIDE + 8 * half];
        const v8bf bhi = *(const v8bf*)&stage[buf][lr * STG_STRIDE + 16 + 8 * half];
        v16bf b;
        #pragma unroll
        for (int i = 0; i < 8; ++i) { b[i] = blo[i]; b[8 + i] = bhi[i]; }
        return b;
    };

    // ---------------- GEMM1: att = sigmoid((data . mem^T) / sqrt(DIM)) ----------------
    {
        v8f acc[16];
        const v8f vzero = {};
        #pragma unroll
        for (int i = 0; i < 16; ++i) acc[i] = vzero;

        const int arow = row0 + strip * 16 + c;
        const float* __restrict__ abase = data + (size_t)arow * DIM;

        // TDM prologue: stage k-chunk 0 (512 rows x 32 bf16 = 512 x 16 dwords)
        if (wave == 0)
            tdm_load_tile(mbf, stg_base[0], 512u, 512u, 16u, 512u, 512u);

        #pragma unroll 1
        for (int kc = 0; kc < DIM / 32; ++kc) {
            const int k   = kc * 32;
            const int buf = kc & 1;

            // A-fragment global loads issued before the handoff so their latency
            // hides under the TDM wait + barrier.
            const float4 f0 = *(const float4*)(abase + k + 8 * half);
            const float4 f1 = *(const float4*)(abase + k + 8 * half + 4);
            const float4 f2 = *(const float4*)(abase + k + 16 + 8 * half);
            const float4 f3 = *(const float4*)(abase + k + 16 + 8 * half + 4);

            if (wave == 0) {
                if (kc + 1 < DIM / 32) {
                    tdm_load_tile(mbf + (size_t)(kc + 1) * 32, stg_base[buf ^ 1],
                                  512u, 512u, 16u, 512u, 512u);
                    __builtin_amdgcn_s_wait_tensorcnt((short)1);  // oldest chunk done
                } else {
                    __builtin_amdgcn_s_wait_tensorcnt((short)0);
                }
            }
            __syncthreads();

            v16bf a;
            a[0]=(bf16_t)f0.x;  a[1]=(bf16_t)f0.y;  a[2]=(bf16_t)f0.z;  a[3]=(bf16_t)f0.w;
            a[4]=(bf16_t)f1.x;  a[5]=(bf16_t)f1.y;  a[6]=(bf16_t)f1.z;  a[7]=(bf16_t)f1.w;
            a[8]=(bf16_t)f2.x;  a[9]=(bf16_t)f2.y;  a[10]=(bf16_t)f2.z; a[11]=(bf16_t)f2.w;
            a[12]=(bf16_t)f3.x; a[13]=(bf16_t)f3.y; a[14]=(bf16_t)f3.z; a[15]=(bf16_t)f3.w;

            // register double-buffer the B fragment: nc+1 loads issue before wmma(nc)
            v16bf bcur = load_bfrag(buf, 0);
            #pragma unroll
            for (int nc = 0; nc < 16; ++nc) {
                v16bf bnext;
                if (nc + 1 < 16) bnext = load_bfrag(buf, nc + 1);
                acc[nc] = __builtin_amdgcn_wmma_f32_16x16x32_bf16(
                    false, a, false, bcur, (short)0, acc[nc], false, false);
                bcur = bnext;
            }
            __syncthreads();
        }

        // sigmoid(x / 32) then store bf16 attention tile to LDS
        #pragma unroll
        for (int nc = 0; nc < 16; ++nc) {
            #pragma unroll
            for (int v = 0; v < 8; ++v) {
                float x = acc[nc][v] * 0.03125f;         // 1/sqrt(1024)
                float s = 1.0f / (1.0f + __expf(-x));
                att[strip * 16 + v + 8 * half][nsel * 256 + nc * 16 + c] = (bf16_t)s;
            }
        }
    }

    __syncthreads();

    // ---------------- temporal_att: mean of top-33 per row ----------------
    {
        #pragma unroll 1
        for (int rr = 0; rr < 8; ++rr) {
            const int lr = wave * 8 + rr;
            float vals[16];
            const v16bf pv = *(const v16bf*)&att[lr][lane * 16];
            #pragma unroll
            for (int i = 0; i < 16; ++i) vals[i] = (float)pv[i];

            float lo = 0.0f, hi = 1.0f;
            #pragma unroll 1
            for (int it = 0; it < 28; ++it) {
                const float mid = 0.5f * (lo + hi);
                int cnt = 0;
                #pragma unroll
                for (int i = 0; i < 16; ++i) cnt += (vals[i] > mid) ? 1 : 0;
                #pragma unroll
                for (int off = 16; off > 0; off >>= 1) cnt += __shfl_xor(cnt, off, 32);
                if (cnt >= KTOP) lo = mid; else hi = mid;
            }
            float s = 0.0f; int cnt = 0;
            #pragma unroll
            for (int i = 0; i < 16; ++i)
                if (vals[i] > hi) { s += vals[i]; ++cnt; }
            #pragma unroll
            for (int off = 16; off > 0; off >>= 1) {
                s   += __shfl_xor(s, off, 32);
                cnt += __shfl_xor(cnt, off, 32);
            }
            if (lane == 0)
                out_t[row0 + lr] = (s + (float)(KTOP - cnt) * lo) * (1.0f / (float)KTOP);
        }
    }
    // top-k only reads att; stage buffers free for reuse

    // ---------------- GEMM2: augment = att . mem ----------------
    const int tid4 = tid >> 2;      // 0..63
    const int part = tid & 3;       // 16B segment within a 64B staged row

    #pragma unroll 1
    for (int pass = 0; pass < 2; ++pass) {
        const int colbase = nsel * 512 + pass * 256;
        v8f acc[16];
        const v8f vzero = {};
        #pragma unroll
        for (int i = 0; i < 16; ++i) acc[i] = vzero;

        // async-stage chunk kc of mbfT: local rows 0..255 <- global rows pass*256..,
        // local rows 256..511 <- global rows 512+pass*256..
        auto issue_stage = [&](int kc) {
            const int buf = kc & 1;
            #pragma unroll
            for (int r = 0; r < 8; ++r) {
                const int lr   = r * 64 + tid4;                      // 0..511
                const int grow = lr + ((lr >> 8) << 8) + pass * 256; // +256 if lr>=256
                const bf16_t* src = mbfT + (size_t)grow * NUMS + kc * 32 + part * 8;
                const uint32_t dst = stg_base[buf] + (uint32_t)(lr * (STG_STRIDE * 2) + part * 16);
                asm volatile("global_load_async_to_lds_b128 %0, %1, off"
                             :: "v"(dst), "v"((uint64_t)(uintptr_t)src)
                             : "memory");
            }
        };

        issue_stage(0);

        const int arow = strip * 16 + c;
        #pragma unroll 1
        for (int kc = 0; kc < NUMS / 32; ++kc) {
            const int k   = kc * 32;
            const int buf = kc & 1;

            // A fragment from the attention tile: untouched by staging, so load
            // before the async wait + barrier to hide LDS latency.
            const v8bf alo = *(const v8bf*)&att[arow][k + 8 * half];
            const v8bf ahi = *(const v8bf*)&att[arow][k + 16 + 8 * half];

            if (kc + 1 < NUMS / 32) {
                issue_stage(kc + 1);
                asm volatile("s_wait_asynccnt 0x8" ::: "memory");   // oldest 8 done
            } else {
                asm volatile("s_wait_asynccnt 0x0" ::: "memory");
            }
            __syncthreads();

            v16bf a;
            #pragma unroll
            for (int i = 0; i < 8; ++i) { a[i] = alo[i]; a[8 + i] = ahi[i]; }

            v16bf bcur = load_bfrag(buf, 0);
            #pragma unroll
            for (int nc = 0; nc < 16; ++nc) {
                v16bf bnext;
                if (nc + 1 < 16) bnext = load_bfrag(buf, nc + 1);
                acc[nc] = __builtin_amdgcn_wmma_f32_16x16x32_bf16(
                    false, a, false, bcur, (short)0, acc[nc], false, false);
                bcur = bnext;
            }
            __syncthreads();
        }

        #pragma unroll
        for (int nc = 0; nc < 16; ++nc) {
            const int col = colbase + nc * 16 + c;
            #pragma unroll
            for (int v = 0; v < 8; ++v) {
                const int row = row0 + strip * 16 + v + 8 * half;
                out_aug[(size_t)row * DIM + col] = acc[nc][v];
            }
        }
    }
}

extern "C" void kernel_launch(void* const* d_in, const int* in_sizes, int n_in,
                              void* d_out, int out_size, void* d_ws, size_t ws_size,
                              hipStream_t stream) {
    (void)in_sizes; (void)n_in; (void)out_size; (void)ws_size;
    const float* data = (const float*)d_in[0];     // [16, 2048, 1024] f32
    const float* mem  = (const float*)d_in[1];     // [512, 1024] f32

    bf16_t* mbf  = (bf16_t*)d_ws;                  // [512 x 1024] bf16
    bf16_t* mbfT = mbf + (size_t)NUMS * DIM;       // [1024 x 512] bf16

    float* out_t   = (float*)d_out;                // [32768] temporal_att
    float* out_aug = out_t + MROWS;                // [32768 x 1024] augment

    memunit_prep<<<(NUMS * DIM) / 256, 256, 0, stream>>>(mem, mbf, mbfT);
    memunit_main<<<MROWS / 64, 256, 0, stream>>>(data, mbf, mbfT, out_t, out_aug);
}
